// AlexNet_CIFAR10_SNN_78812649881689
// MI455X (gfx1250) — matrix operand order repaired and checked by reading, not compiled, not run on hardware
//
#include <hip/hip_runtime.h>
#include <stdint.h>

// ---------------------------------------------------------------------------
// AlexNet-CIFAR10 SNN inference, T=12, B=256, for MI455X (gfx1250, wave32).
// Heavy layers (conv2..conv5, fc1, fc2) are bf16 WMMA implicit GEMMs (spikes
// are exactly {0,1} in bf16), with double-buffered async global->LDS staging
// (GLOBAL_LOAD_ASYNC_TO_LDS_B128, ASYNCcnt pipelining). IF membrane scan
// (+fused 2x2 maxpool = OR of binary spikes) are elementwise kernels. conv1
// computed once (T-invariant input). fc3 + temporal mean is a small reduction.
// ---------------------------------------------------------------------------

typedef unsigned short ushort_t;
typedef __attribute__((ext_vector_type(16))) __bf16 v16bf;
typedef __attribute__((ext_vector_type(8)))  float  v8f;
typedef __attribute__((ext_vector_type(4)))  int    i32x4;

#define T_STEPS 12
#define BATCH   256

#define AS1 __attribute__((address_space(1)))
#define AS3 __attribute__((address_space(3)))

#if __has_builtin(__builtin_amdgcn_global_load_async_to_lds_b128)
#define HAVE_ASYNC_LDS 1
#else
#define HAVE_ASYNC_LDS 0
#endif

#if HAVE_ASYNC_LDS
static __device__ __forceinline__ void wait_async_le0() {
#if __has_builtin(__builtin_amdgcn_s_wait_asynccnt)
  __builtin_amdgcn_s_wait_asynccnt(0);
#else
  asm volatile("s_wait_asynccnt 0x0" ::: "memory");
#endif
}
static __device__ __forceinline__ void wait_async_le6() {
#if __has_builtin(__builtin_amdgcn_s_wait_asynccnt)
  __builtin_amdgcn_s_wait_asynccnt(6);
#else
  asm volatile("s_wait_asynccnt 0x6" ::: "memory");
#endif
}
#endif

// ---- bf16 helpers (raw ushort storage) ------------------------------------
static __device__ __forceinline__ ushort_t f2bf(float f) {
  unsigned u = __float_as_uint(f);
  unsigned r = (u + 0x7FFFu + ((u >> 16) & 1u)) >> 16;   // RNE
  return (ushort_t)r;
}
static __device__ __forceinline__ float bf2f(ushort_t u) {
  return __uint_as_float(((unsigned)u) << 16);
}

// ===========================================================================
// WMMA implicit-GEMM conv / fc kernel (bf16 A/B, fp32 accum, +bias)
//   A : spikes, NHWC  [IMGS, H, W, Cin]  (bf16 raw)
//   Wt: weights       [KH][KW][Cout][Cin] (bf16 raw)  -- N-major, K contiguous
//   Zp: 16B (or more) of global zeros -- async source for padded border rows
//   Y : output, NHWC  [IMGS, H, W, Cout] (fp32)
// Block tile 128(M) x 64(N); K-block 64 (two K=32 WMMA steps, 8 v_wmma);
// double-buffered LDS, all staging via async global->LDS (ASYNCcnt),
// issue(k+1) overlaps compute(k).
// Requires: M %128==0, Cout %64==0, Cin %64==0 (true for all layers used).
// ===========================================================================
#define BM 128
#define BN 64
#define KB 64    // K elements staged in LDS per block
#define LDA 72   // padded LDS row stride (elements); 144B rows keep b128 aligned
#define LDB 72

__global__ __launch_bounds__(256)
void snn_gemm_bf16(const ushort_t* __restrict__ A,
                   const ushort_t* __restrict__ Wt,
                   const float* __restrict__ bias,
                   const ushort_t* __restrict__ Zp,
                   float* __restrict__ Y,
                   int H, int Wd, int Cin, int Cout,
                   int KH, int KW, int pad)
{
  __shared__ ushort_t As[2][BM * LDA];
  __shared__ ushort_t Bs[2][BN * LDB];

  const int tid  = threadIdx.x;
  const int lane = tid & 31;
  const int wave = tid >> 5;        // 0..7
  const int wm   = wave >> 1;       // 0..3  -> 32-row strip
  const int wn   = wave & 1;        // 0..1  -> 32-col strip
  const int m0   = blockIdx.y * BM;
  const int n0   = blockIdx.x * BN;

  // Pre-decode the 4 A-tile rows this thread services (row = idx>>3).
  int r_img[4], r_h[4], r_w[4];
#pragma unroll
  for (int it = 0; it < 4; ++it) {
    int idx = it * 256 + tid;
    int row = idx >> 3;             // 0..127
    int m   = m0 + row;
    int wq  = m % Wd;  int tv = m / Wd;
    int hq  = tv % H;  int img = tv / H;
    r_img[it] = img; r_h[it] = hq; r_w[it] = wq;
  }

  v8f acc[2][2] = {};

  // 8 WMMAs (two K=32 steps) from LDS buffer `buf` (k = ks + 16*half + s in both A and B).
  auto compute = [&](int buf) {
    const int half = lane >> 4;
    const int lrow = lane & 15;
    union Frag { uint4 q[2]; v16bf v; };
#pragma unroll
    for (int ks = 0; ks < KB; ks += 32) {
      Frag fa[2], fb[2];
#pragma unroll
      for (int tm = 0; tm < 2; ++tm) {
        int r = wm * 32 + tm * 16 + lrow;       // A row (M)
        fa[tm].q[0] = *(const uint4*)(&As[buf][r * LDA + ks + half * 16]);
        fa[tm].q[1] = *(const uint4*)(&As[buf][r * LDA + ks + half * 16 + 8]);
      }
#pragma unroll
      for (int tn = 0; tn < 2; ++tn) {
        int r = wn * 32 + tn * 16 + lrow;       // B column (N)
        fb[tn].q[0] = *(const uint4*)(&Bs[buf][r * LDB + ks + half * 16]);
        fb[tn].q[1] = *(const uint4*)(&Bs[buf][r * LDB + ks + half * 16 + 8]);
      }
#pragma unroll
      for (int tm = 0; tm < 2; ++tm)
#pragma unroll
        for (int tn = 0; tn < 2; ++tn)
          acc[tm][tn] = __builtin_amdgcn_wmma_f32_16x16x32_bf16(
              false, fa[tm].v, false, fb[tn].v,
              (short)0, acc[tm][tn], false, false);
    }
  };

  const int nb = KH * KW * (Cin / KB);      // total K-blocks
  int kh_i = 0, kw_i = 0, kc_i = 0;         // issue cursor
  auto advance = [&]() {
    kc_i += KB;
    if (kc_i == Cin) { kc_i = 0; kw_i++; if (kw_i == KW) { kw_i = 0; kh_i++; } }
  };

#if HAVE_ASYNC_LDS
  // Issue one K-block's 6 async b128 transfers (4 A + 2 B) into buffer `buf`.
  // Border rows source from the global zero block (address select, no VGPR
  // data path) so the whole tile moves memory->LDS asynchronously.
  auto issue = [&](int buf) {
#pragma unroll
    for (int it = 0; it < 4; ++it) {
      int idx = it * 256 + tid;
      int row = idx >> 3;
      int c8  = (idx & 7) << 3;               // 0,8,...,56
      int ih  = r_h[it] + kh_i - pad;
      int iw  = r_w[it] + kw_i - pad;
      bool valid = ((unsigned)ih < (unsigned)H) & ((unsigned)iw < (unsigned)Wd);
      const ushort_t* gsrc = valid
          ? A + ((size_t)(r_img[it] * H + ih) * Wd + iw) * Cin + kc_i + c8
          : Zp;
      __builtin_amdgcn_global_load_async_to_lds_b128(
          (AS1 i32x4*)gsrc, (AS3 i32x4*)(&As[buf][row * LDA + c8]), 0, 0);
    }
    const ushort_t* wslice =
        Wt + ((size_t)(kh_i * KW + kw_i) * Cout + n0) * Cin + kc_i;
#pragma unroll
    for (int it = 0; it < 2; ++it) {
      int idx = it * 256 + tid;
      int row = idx >> 3;                     // n: 0..63
      int c8  = (idx & 7) << 3;
      __builtin_amdgcn_global_load_async_to_lds_b128(
          (AS1 i32x4*)(wslice + (size_t)row * Cin + c8),
          (AS3 i32x4*)(&Bs[buf][row * LDB + c8]), 0, 0);
    }
  };

  // Double-buffered pipeline: issue(k+1) overlaps compute(k).
  issue(0); advance();
  for (int bi = 0; bi < nb; ++bi) {
    int cur = bi & 1;
    if (bi + 1 < nb) {
      issue((bi + 1) & 1); advance();
      wait_async_le6();                       // block bi resident (in-order)
    } else {
      wait_async_le0();
    }
    __syncthreads();
    compute(cur);
    __syncthreads();                          // reads done before buf reuse
  }
#else
  // Fallback: synchronous staging with clamp+select boundary masking.
  for (int bi = 0; bi < nb; ++bi) {
#pragma unroll
    for (int it = 0; it < 4; ++it) {
      int idx = it * 256 + tid;
      int row = idx >> 3;
      int c8  = (idx & 7) << 3;
      int ih  = r_h[it] + kh_i - pad;
      int iw  = r_w[it] + kw_i - pad;
      bool valid = ((unsigned)ih < (unsigned)H) & ((unsigned)iw < (unsigned)Wd);
      int ihc = valid ? ih : 0;
      int iwc = valid ? iw : 0;
      size_t src = ((size_t)(r_img[it] * H + ihc) * Wd + iwc) * Cin + kc_i + c8;
      uint4 v = *(const uint4*)(A + src);
      if (!valid) { v.x = 0u; v.y = 0u; v.z = 0u; v.w = 0u; }
      *(uint4*)(&As[0][row * LDA + c8]) = v;
    }
    const ushort_t* wslice =
        Wt + ((size_t)(kh_i * KW + kw_i) * Cout + n0) * Cin + kc_i;
#pragma unroll
    for (int it = 0; it < 2; ++it) {
      int idx = it * 256 + tid;
      int row = idx >> 3;
      int c8  = (idx & 7) << 3;
      uint4 v = *(const uint4*)(wslice + (size_t)row * Cin + c8);
      *(uint4*)(&Bs[0][row * LDB + c8]) = v;
    }
    __syncthreads();
    compute(0);
    __syncthreads();
    advance();
  }
#endif

  // ---- store D (f32 16x16 layout: M = r + 8*half, N = lane&15) + bias ----
  const int half = lane >> 4;
  const int nn   = lane & 15;
#pragma unroll
  for (int tm = 0; tm < 2; ++tm)
#pragma unroll
    for (int tn = 0; tn < 2; ++tn) {
      int gn = n0 + wn * 32 + tn * 16 + nn;
      float bv = bias[gn];
#pragma unroll
      for (int r = 0; r < 8; ++r) {
        int gm = m0 + wm * 32 + tm * 16 + half * 8 + r;
        Y[(size_t)gm * Cout + gn] = acc[tm][tn][r] + bv;
      }
    }
}

// ===========================================================================
// IF membrane scan (hard reset), spikes out as bf16 {0,1}.
// ===========================================================================
__global__ void snn_if_seq(const float* __restrict__ Y, ushort_t* __restrict__ S,
                           int R, long long strideT)
{
  int j = blockIdx.x * blockDim.x + threadIdx.x;
  if (j >= R) return;
  float v = 0.0f;
  for (int t = 0; t < T_STEPS; ++t) {
    float h = v + Y[(size_t)t * strideT + j];
    bool s = (h >= 1.0f);
    S[(size_t)t * R + j] = s ? (ushort_t)0x3F80 : (ushort_t)0;
    v = s ? 0.0f : h;
  }
}

// IF scan fused with 2x2 maxpool (pool of binary spikes = OR).
__global__ void snn_if_pool(const float* __restrict__ Y, ushort_t* __restrict__ S,
                            int Bn, int H, int Wd, int C, long long ystrideT)
{
  const int PH = H >> 1, PW = Wd >> 1;
  const int Rout = Bn * PH * PW * C;
  int j = blockIdx.x * blockDim.x + threadIdx.x;
  if (j >= Rout) return;
  int c  = j % C;   int tv = j / C;
  int pw = tv % PW; tv /= PW;
  int ph = tv % PH; int b = tv / PH;
  size_t i00 = ((size_t)(b * H + 2 * ph) * Wd + 2 * pw) * C + c;
  size_t i01 = i00 + C;
  size_t i10 = i00 + (size_t)Wd * C;
  size_t i11 = i10 + C;
  float v0 = 0.f, v1 = 0.f, v2 = 0.f, v3 = 0.f;
  for (int t = 0; t < T_STEPS; ++t) {
    size_t base = (size_t)t * ystrideT;
    float h0 = v0 + Y[base + i00]; bool s0 = h0 >= 1.0f; v0 = s0 ? 0.f : h0;
    float h1 = v1 + Y[base + i01]; bool s1 = h1 >= 1.0f; v1 = s1 ? 0.f : h1;
    float h2 = v2 + Y[base + i10]; bool s2 = h2 >= 1.0f; v2 = s2 ? 0.f : h2;
    float h3 = v3 + Y[base + i11]; bool s3 = h3 >= 1.0f; v3 = s3 ? 0.f : h3;
    S[(size_t)t * Rout + j] = (s0 | s1 | s2 | s3) ? (ushort_t)0x3F80 : (ushort_t)0;
  }
}

// ===========================================================================
// conv1: 3->64, 3x3 stride2 pad1, fp32 direct (input is T-invariant)
// X: [B,3,32,32] NCHW, Y1: [B,16,16,64] NHWC
// ===========================================================================
__global__ void snn_conv1(const float* __restrict__ X, const float* __restrict__ W1,
                          const float* __restrict__ B1, float* __restrict__ Y1)
{
  int j = blockIdx.x * blockDim.x + threadIdx.x;        // B*16*16*64
  if (j >= BATCH * 16 * 16 * 64) return;
  int co = j & 63; int tv = j >> 6;
  int w = tv & 15; tv >>= 4;
  int h = tv & 15; int b = tv >> 4;
  float acc = B1[co];
#pragma unroll
  for (int ci = 0; ci < 3; ++ci)
#pragma unroll
    for (int kh = 0; kh < 3; ++kh) {
      int ih = 2 * h + kh - 1;
      if (ih < 0 || ih > 31) continue;
#pragma unroll
      for (int kw = 0; kw < 3; ++kw) {
        int iw = 2 * w + kw - 1;
        if (iw < 0 || iw > 31) continue;
        acc += X[((size_t)(b * 3 + ci) * 32 + ih) * 32 + iw]
             * W1[((size_t)(co * 3 + ci) * 3 + kh) * 3 + kw];
      }
    }
  Y1[j] = acc;
}

// ===========================================================================
// Weight prep: OIHW fp32 -> [kh][kw][O][I] bf16   (N-major, K contiguous)
// ===========================================================================
__global__ void snn_prep_convw(const float* __restrict__ Ws, ushort_t* __restrict__ Wd,
                               int O, int I)
{
  int j = blockIdx.x * blockDim.x + threadIdx.x;
  int total = 9 * O * I;
  if (j >= total) return;
  int i = j % I; int tv = j / I;
  int o = tv % O; int kk = tv / O;        // kk = kh*3+kw
  int kh = kk / 3, kw = kk % 3;
  Wd[j] = f2bf(Ws[((size_t)(o * I + i) * 3 + kh) * 3 + kw]);
}

// generic fp32 -> bf16 convert, layout preserved ([N][K])
__global__ void snn_prep_fcw(const float* __restrict__ s, ushort_t* __restrict__ d, int n)
{
  int j = blockIdx.x * blockDim.x + threadIdx.x;
  if (j < n) d[j] = f2bf(s[j]);
}

// fw1: permute K from NCHW-flatten (c*4+h*2+w) to NHWC-flatten ((h*2+w)*256+c)
__global__ void snn_prep_fw1(const float* __restrict__ s, ushort_t* __restrict__ d)
{
  int j = blockIdx.x * blockDim.x + threadIdx.x;       // 4096*1024
  if (j >= 4096 * 1024) return;
  int i_out = j & 1023; int o = j >> 10;
  int hw = i_out >> 8;  int c = i_out & 255;
  d[j] = f2bf(s[(size_t)o * 1024 + c * 4 + hw]);
}

// zero the border-source block
__global__ void snn_zero128(ushort_t* __restrict__ z)
{
  if (threadIdx.x < 128) z[threadIdx.x] = 0;
}

// ===========================================================================
// fc3 + temporal mean: out[b,o] = fb3[o] + (1/T) * sum_t sum_k s7[t,b,k]*fw3[o,k]
// ===========================================================================
__global__ __launch_bounds__(256)
void snn_fc3_mean(const ushort_t* __restrict__ S, const float* __restrict__ FW3,
                  const float* __restrict__ FB3, float* __restrict__ OUT, int K)
{
  const int b = blockIdx.x / 10;
  const int o = blockIdx.x % 10;
  __shared__ float red[256];
  float acc = 0.0f;
  const float* wrow = FW3 + (size_t)o * K;
  for (int t = 0; t < T_STEPS; ++t) {
    const ushort_t* srow = S + ((size_t)t * BATCH + b) * K;
    for (int k = threadIdx.x; k < K; k += 256)
      acc += bf2f(srow[k]) * wrow[k];
  }
  red[threadIdx.x] = acc;
  __syncthreads();
  for (int s = 128; s > 0; s >>= 1) {
    if (threadIdx.x < s) red[threadIdx.x] += red[threadIdx.x + s];
    __syncthreads();
  }
  if (threadIdx.x == 0)
    OUT[b * 10 + o] = red[0] * (1.0f / (float)T_STEPS) + FB3[o];
}

// ===========================================================================
// Workspace layout (bytes, 256-aligned). ~260 MB total.
// ===========================================================================
static const size_t OFF_W2  = 0;                         // 9*192*64*2   = 221184
static const size_t OFF_W3  = 221184;                    // 9*384*192*2  = 1327104
static const size_t OFF_W4  = 1548288;                   // 9*256*384*2  = 1769472
static const size_t OFF_W5  = 3317760;                   // 9*256*256*2  = 1179648
static const size_t OFF_FW1 = 4497408;                   // 4096*1024*2  = 8388608
static const size_t OFF_FW2 = 12886016;                  // 4096*4096*2  = 33554432
static const size_t OFF_Y   = 46440448;                  // 3072*64*192*4 = 150994944 (max)
static const size_t OFF_S0  = 197435392;                 // 37748736 (max of s1,s3,s5,s7)
static const size_t OFF_S1  = 235184128;                 // 25165824 (max of s2,s4,s6)
static const size_t OFF_Z   = 260349952;                 // 256B zero block

extern "C" void kernel_launch(void* const* d_in, const int* in_sizes, int n_in,
                              void* d_out, int out_size, void* d_ws, size_t ws_size,
                              hipStream_t stream) {
  const float* x   = (const float*)d_in[0];
  const float* w1  = (const float*)d_in[1];  const float* b1  = (const float*)d_in[2];
  const float* w2  = (const float*)d_in[3];  const float* b2  = (const float*)d_in[4];
  const float* w3  = (const float*)d_in[5];  const float* b3  = (const float*)d_in[6];
  const float* w4  = (const float*)d_in[7];  const float* b4  = (const float*)d_in[8];
  const float* w5  = (const float*)d_in[9];  const float* b5  = (const float*)d_in[10];
  const float* fw1 = (const float*)d_in[11]; const float* fb1 = (const float*)d_in[12];
  const float* fw2 = (const float*)d_in[13]; const float* fb2 = (const float*)d_in[14];
  const float* fw3 = (const float*)d_in[15]; const float* fb3 = (const float*)d_in[16];
  float* out = (float*)d_out;

  uint8_t* ws = (uint8_t*)d_ws;
  ushort_t* w2t  = (ushort_t*)(ws + OFF_W2);
  ushort_t* w3t  = (ushort_t*)(ws + OFF_W3);
  ushort_t* w4t  = (ushort_t*)(ws + OFF_W4);
  ushort_t* w5t  = (ushort_t*)(ws + OFF_W5);
  ushort_t* fw1t = (ushort_t*)(ws + OFF_FW1);
  ushort_t* fw2t = (ushort_t*)(ws + OFF_FW2);
  float*    Y    = (float*)(ws + OFF_Y);
  ushort_t* S0   = (ushort_t*)(ws + OFF_S0);
  ushort_t* S1   = (ushort_t*)(ws + OFF_S1);
  ushort_t* Zp   = (ushort_t*)(ws + OFF_Z);

  const int TB = T_STEPS * BATCH;        // 3072
  auto blocks = [](long long n, int bs) { return (unsigned)((n + bs - 1) / bs); };

  // ---- prep (deterministic each call) ----
  snn_zero128<<<1, 128, 0, stream>>>(Zp);
  snn_prep_convw<<<blocks(9LL*192*64, 256), 256, 0, stream>>>(w2, w2t, 192, 64);
  snn_prep_convw<<<blocks(9LL*384*192, 256), 256, 0, stream>>>(w3, w3t, 384, 192);
  snn_prep_convw<<<blocks(9LL*256*384, 256), 256, 0, stream>>>(w4, w4t, 256, 384);
  snn_prep_convw<<<blocks(9LL*256*256, 256), 256, 0, stream>>>(w5, w5t, 256, 256);
  snn_prep_fw1 <<<blocks(4096LL*1024, 256), 256, 0, stream>>>(fw1, fw1t);
  snn_prep_fcw <<<blocks(4096LL*4096, 256), 256, 0, stream>>>(fw2, fw2t, 4096*4096);

  // ---- layer 1: conv (once, T-invariant) -> IF+pool -> s1 [T,B,8,8,64] ----
  snn_conv1<<<blocks((long long)BATCH*16*16*64, 256), 256, 0, stream>>>(x, w1, b1, Y);
  snn_if_pool<<<blocks((long long)BATCH*8*8*64, 256), 256, 0, stream>>>(
      Y, S0, BATCH, 16, 16, 64, /*ystrideT=*/0);

  // ---- conv2: [TB,8,8,64] -> [TB,8,8,192]; IF+pool -> s2 [T,B,4,4,192] ----
  snn_gemm_bf16<<<dim3(192/BN, (unsigned)((long long)TB*64/BM)), 256, 0, stream>>>(
      S0, w2t, b2, Zp, Y, 8, 8, 64, 192, 3, 3, 1);
  snn_if_pool<<<blocks((long long)BATCH*4*4*192, 256), 256, 0, stream>>>(
      Y, S1, BATCH, 8, 8, 192, (long long)BATCH*8*8*192);

  // ---- conv3: -> [TB,4,4,384]; IF -> s3 ----
  snn_gemm_bf16<<<dim3(384/BN, (unsigned)((long long)TB*16/BM)), 256, 0, stream>>>(
      S1, w3t, b3, Zp, Y, 4, 4, 192, 384, 3, 3, 1);
  snn_if_seq<<<blocks((long long)BATCH*16*384, 256), 256, 0, stream>>>(
      Y, S0, BATCH*16*384, (long long)BATCH*16*384);

  // ---- conv4: -> [TB,4,4,256]; IF -> s4 ----
  snn_gemm_bf16<<<dim3(256/BN, (unsigned)((long long)TB*16/BM)), 256, 0, stream>>>(
      S0, w4t, b4, Zp, Y, 4, 4, 384, 256, 3, 3, 1);
  snn_if_seq<<<blocks((long long)BATCH*16*256, 256), 256, 0, stream>>>(
      Y, S1, BATCH*16*256, (long long)BATCH*16*256);

  // ---- conv5: -> [TB,4,4,256]; IF+pool -> s5 [T,B,2,2,256]=[T,B,1024] ----
  snn_gemm_bf16<<<dim3(256/BN, (unsigned)((long long)TB*16/BM)), 256, 0, stream>>>(
      S1, w5t, b5, Zp, Y, 4, 4, 256, 256, 3, 3, 1);
  snn_if_pool<<<blocks((long long)BATCH*2*2*256, 256), 256, 0, stream>>>(
      Y, S0, BATCH, 4, 4, 256, (long long)BATCH*4*4*256);

  // ---- fc1: [TB,1024] -> [TB,4096] (1x1 "conv"); IF -> s6 ----
  snn_gemm_bf16<<<dim3(4096/BN, (unsigned)(TB/BM)), 256, 0, stream>>>(
      S0, fw1t, fb1, Zp, Y, 1, 1, 1024, 4096, 1, 1, 0);
  snn_if_seq<<<blocks((long long)BATCH*4096, 256), 256, 0, stream>>>(
      Y, S1, BATCH*4096, (long long)BATCH*4096);

  // ---- fc2: [TB,4096] -> [TB,4096]; IF -> s7 ----
  snn_gemm_bf16<<<dim3(4096/BN, (unsigned)(TB/BM)), 256, 0, stream>>>(
      S1, fw2t, fb2, Zp, Y, 1, 1, 4096, 4096, 1, 1, 0);
  snn_if_seq<<<blocks((long long)BATCH*4096, 256), 256, 0, stream>>>(
      Y, S0, BATCH*4096, (long long)BATCH*4096);

  // ---- fc3 + temporal mean -> out [256,10] ----
  snn_fc3_mean<<<BATCH * 10, 256, 0, stream>>>(S0, fw3, fb3, out, 4096);
}
